// StarTransformerClassifier_43430709297361
// MI455X (gfx1250) — compile-verified, matrix operand order checked by reading, not run on hardware
//
#include <hip/hip_runtime.h>
#include <hip/hip_bf16.h>
#include <math.h>

// ---------------------------------------------------------------------------
// Problem constants (match the reference)
// ---------------------------------------------------------------------------
#define SEQ    2048
#define DMODEL 1536
#define NSPAN  512
#define KNEI   16
#define FFD    2048

typedef __attribute__((ext_vector_type(16))) __bf16 v16bf;
typedef __attribute__((ext_vector_type(8)))  float  v8f;
typedef __attribute__((ext_vector_type(4)))  unsigned int v4u;
typedef __attribute__((ext_vector_type(8)))  int    v8i;
typedef __attribute__((ext_vector_type(4)))  int    v4i;

union FragB16 {
    v16bf v;
    unsigned int u[8];
};

__device__ __forceinline__ unsigned int pack_bf16(float a, float b) {
    unsigned int ua = __float_as_uint(a);
    unsigned int ub = __float_as_uint(b);
    ua = (ua + 0x7FFFu + ((ua >> 16) & 1u)) >> 16;   // RNE f32 -> bf16
    ub = (ub + 0x7FFFu + ((ub >> 16) & 1u)) >> 16;
    return (ua & 0xFFFFu) | (ub << 16);
}

__device__ __forceinline__ unsigned short cvt_bf16(float a) {
    unsigned int ua = __float_as_uint(a);
    ua = (ua + 0x7FFFu + ((ua >> 16) & 1u)) >> 16;
    return (unsigned short)ua;
}

// ---------------------------------------------------------------------------
// TDM: async 2D bf16 tile load (tile_w elems x tile_h rows, row stride =
// row_stride elems) from global into LDS at lds_off. Descriptor per
// CDNA5 ISA (D# group0/group1; groups 2/3 zero).
// Issue from ONE wave only; follow with s_wait_tensorcnt before use.
// ---------------------------------------------------------------------------
__device__ __forceinline__ void tdm_load_tile_bf16(unsigned lds_off,
                                                   const unsigned short* gptr,
                                                   unsigned tile_w, unsigned tile_h,
                                                   unsigned row_stride) {
    unsigned long long ga = (unsigned long long)gptr;   // byte address of tile start
    v4u g0;
    g0[0] = 1u;                                          // count=1, user descriptor
    g0[1] = lds_off;                                     // LDS byte address
    g0[2] = (unsigned)(ga & 0xFFFFFFFFu);                // global_addr[31:0]
    g0[3] = (unsigned)((ga >> 32) & 0x01FFFFFFu) | (2u << 30);  // addr[56:32] | type=2

    unsigned long long s0 = (unsigned long long)row_stride;     // dim0 stride (elems)
    v8i g1;
    g1[0] = (int)(1u << 16);                             // data_size=1 (2 bytes)
    g1[1] = (int)((tile_w & 0xFFFFu) << 16);             // tensor_dim0[15:0] @bits63:48
    g1[2] = (int)(((tile_w >> 16) & 0xFFFFu) | ((tile_h & 0xFFFFu) << 16)); // td0 hi | td1 lo
    g1[3] = (int)(((tile_h >> 16) & 0xFFFFu) | ((tile_w & 0xFFFFu) << 16)); // td1 hi | tile_dim0
    g1[4] = (int)(tile_h & 0xFFFFu);                     // tile_dim1 (tile_dim2=0)
    g1[5] = (int)(unsigned)(s0 & 0xFFFFFFFFu);           // tensor_dim0_stride[31:0]
    g1[6] = (int)(unsigned)((s0 >> 32) & 0xFFFFu);       // stride[47:32] (stride1=0)
    g1[7] = 0;

    v4i z4 = {0, 0, 0, 0};
#if defined(__clang_major__) && (__clang_major__ >= 23)
    v8i z8 = {0, 0, 0, 0, 0, 0, 0, 0};
    __builtin_amdgcn_tensor_load_to_lds(g0, g1, z4, z4, z8, 0);
#else
    __builtin_amdgcn_tensor_load_to_lds(g0, g1, z4, z4, 0);
#endif
}

// ---------------------------------------------------------------------------
// f32 -> bf16 bulk conversion (one pass per operand; n2 = element_count/2)
// ---------------------------------------------------------------------------
__global__ __launch_bounds__(256)
void f32_to_bf16_kernel(const float* __restrict__ src, unsigned int* __restrict__ dst,
                        long n2) {
    long i = (long)blockIdx.x * 256 + threadIdx.x;
    if (i < n2) {
        const float2 f = ((const float2*)src)[i];
        dst[i] = pack_bf16(f.x, f.y);
    }
}

// ---------------------------------------------------------------------------
// Generic GEMM on bf16 operands, fp32 accumulation:
//   C[M,N] = relu?( alpha * A[M,K] x B(K,N | N,K if TB) + bias + res )
// Outputs: optional f32 C and/or bf16 Ch.
// 256 threads = 8 wave32, 128x128 tile, each wave 32x64 (2x4 WMMA tiles).
// A tile (and B tile when TB) staged LDS via the Tensor Data Mover.
// ---------------------------------------------------------------------------
template <bool TB>
__global__ __launch_bounds__(256)
void gemm_bf16_wmma(const unsigned short* __restrict__ A,
                    const unsigned short* __restrict__ B,
                    const float* __restrict__ bias, const float* __restrict__ res,
                    float* __restrict__ C, unsigned short* __restrict__ Ch,
                    int M, int N, int K, float alpha, int relu) {
    __shared__ unsigned int lA[128 * 16];   // 128 rows x 32 bf16
    __shared__ unsigned int lB[128 * 16];   // 128 cols x 32 bf16 ([n][k])

    const int tid  = threadIdx.x;
    const int lane = tid & 31;
    const int wid  = tid >> 5;
    const int wm   = wid >> 1;          // 0..3 (M)
    const int wn   = wid & 1;           // 0..1 (N)
    const int hf   = lane >> 4;
    const int lr   = lane & 15;
    const int m0   = blockIdx.y * 128;
    const int n0   = blockIdx.x * 128;

    const unsigned ldsA_off = (unsigned)(unsigned long long)(void*)lA;
    const unsigned ldsB_off = (unsigned)(unsigned long long)(void*)lB;

    v8f acc[2][4];
#pragma unroll
    for (int i = 0; i < 2; ++i)
#pragma unroll
        for (int j = 0; j < 4; ++j)
            acc[i][j] = (v8f){0.f, 0.f, 0.f, 0.f, 0.f, 0.f, 0.f, 0.f};

    for (int kb = 0; kb < K; kb += 32) {
        // ---- async TDM stage: A tile (always), B tile (TB case) ----
        if (wid == 0) {
            tdm_load_tile_bf16(ldsA_off, A + (size_t)m0 * K + kb, 32u, 128u, (unsigned)K);
            if (TB)
                tdm_load_tile_bf16(ldsB_off, B + (size_t)n0 * K + kb, 32u, 128u, (unsigned)K);
        }
        if (!TB) {
            // B is K x N bf16: coalesced dword loads over n, 16-bit scatter to [n][k]
            unsigned short* lBh = (unsigned short*)lB;
#pragma unroll
            for (int j = 0; j < 8; ++j) {
                int e  = tid + 256 * j;           // 2048 dwords
                int k  = e >> 6;                  // 0..31
                int n  = (e & 63) * 2;            // 0..126
                unsigned u = *(const unsigned int*)(B + (size_t)(kb + k) * N + n0 + n);
                lBh[n * 32 + k]       = (unsigned short)(u & 0xFFFFu);
                lBh[(n + 1) * 32 + k] = (unsigned short)(u >> 16);
            }
        }
        if (wid == 0) __builtin_amdgcn_s_wait_tensorcnt(0);
        __syncthreads();

        // ---- fragments per gfx1250 16-bit A/B VGPR layout ----
        FragB16 aF[2], bF[4];
#pragma unroll
        for (int mi = 0; mi < 2; ++mi) {
            int row = wm * 32 + mi * 16 + lr;
#pragma unroll
            for (int v = 0; v < 8; ++v) {
                int uidx = hf * 4 + v + ((v >= 4) ? 4 : 0);  // K = hf*8 + {0..7,16..23}
                aF[mi].u[v] = lA[row * 16 + uidx];
            }
        }
#pragma unroll
        for (int ni = 0; ni < 4; ++ni) {
            int col = wn * 64 + ni * 16 + lr;
#pragma unroll
            for (int v = 0; v < 8; ++v) {
                int uidx = hf * 4 + v + ((v >= 4) ? 4 : 0);
                bF[ni].u[v] = lB[col * 16 + uidx];
            }
        }

#pragma unroll
        for (int mi = 0; mi < 2; ++mi)
#pragma unroll
            for (int ni = 0; ni < 4; ++ni)
                acc[mi][ni] = __builtin_amdgcn_wmma_f32_16x16x32_bf16(
                    false, aF[mi].v, false, bF[ni].v, (short)0, acc[mi][ni],
                    false, false);
        __syncthreads();
    }

    // ---- epilogue: lane = column, VGPR r = row (+8 in upper half) ----
#pragma unroll
    for (int mi = 0; mi < 2; ++mi) {
#pragma unroll
        for (int ni = 0; ni < 4; ++ni) {
            int col  = n0 + wn * 64 + ni * 16 + lr;
            int rowb = m0 + wm * 32 + mi * 16 + hf * 8;
            float bv = bias ? bias[col] : 0.0f;
#pragma unroll
            for (int r = 0; r < 8; ++r) {
                int row = rowb + r;
                float v = alpha * acc[mi][ni][r] + bv;
                if (res) v += res[(size_t)row * N + col];
                if (relu) v = fmaxf(v, 0.0f);
                if (C)  C[(size_t)row * N + col] = v;
                if (Ch) Ch[(size_t)row * N + col] = cvt_bf16(v);
            }
        }
    }
}

// ---------------------------------------------------------------------------
// Row softmax over f32, emits bf16 probabilities (f32 buffer is scratch-only)
// ---------------------------------------------------------------------------
__global__ __launch_bounds__(256)
void softmax_rows(float* __restrict__ Mt, unsigned short* __restrict__ Mh, int cols) {
    __shared__ float red[256];
    const int tid = threadIdx.x;
    float* p = Mt + (size_t)blockIdx.x * cols;
    unsigned short* ph = Mh + (size_t)blockIdx.x * cols;

    float mx = -3.402823e38f;
    for (int c = tid; c < cols; c += 256) mx = fmaxf(mx, p[c]);
    red[tid] = mx;
    __syncthreads();
    for (int s = 128; s > 0; s >>= 1) {
        if (tid < s) red[tid] = fmaxf(red[tid], red[tid + s]);
        __syncthreads();
    }
    mx = red[0];
    __syncthreads();

    float sum = 0.f;
    for (int c = tid; c < cols; c += 256) {
        float e = __expf(p[c] - mx);
        p[c] = e;
        sum += e;
    }
    red[tid] = sum;
    __syncthreads();
    for (int s = 128; s > 0; s >>= 1) {
        if (tid < s) red[tid] += red[tid + s];
        __syncthreads();
    }
    float inv = 1.0f / red[0];
    for (int c = tid; c < cols; c += 256) ph[c] = cvt_bf16(p[c] * inv);
}

// ---------------------------------------------------------------------------
// Row LayerNorm with optional f32 and/or bf16 outputs
// ---------------------------------------------------------------------------
__global__ __launch_bounds__(256)
void layernorm_rows(const float* __restrict__ X, const float* __restrict__ g,
                    const float* __restrict__ b, float* __restrict__ Yf,
                    unsigned short* __restrict__ Yh, int cols) {
    __shared__ float rs[256];
    __shared__ float rq[256];
    const int tid = threadIdx.x;
    const float* x = X + (size_t)blockIdx.x * cols;

    float s = 0.f, q = 0.f;
    for (int c = tid; c < cols; c += 256) {
        float v = x[c];
        s += v;
        q += v * v;
    }
    rs[tid] = s;
    rq[tid] = q;
    __syncthreads();
    for (int st = 128; st > 0; st >>= 1) {
        if (tid < st) { rs[tid] += rs[tid + st]; rq[tid] += rq[tid + st]; }
        __syncthreads();
    }
    float mean = rs[0] / (float)cols;
    float var  = rq[0] / (float)cols - mean * mean;
    float inv  = rsqrtf(var + 1e-5f);
    for (int c = tid; c < cols; c += 256) {
        float v = g[c] * (x[c] - mean) * inv + b[c];
        if (Yf) Yf[(size_t)blockIdx.x * cols + c] = v;
        if (Yh) Yh[(size_t)blockIdx.x * cols + c] = cvt_bf16(v);
    }
}

// ---------------------------------------------------------------------------
// Pair kernel: one wave32 per span pair (a,b); bf16 neighbor data.
//   block[i][j] = dot(neighbors[a,i,:], embs_neighbors[b,j,:])  via 48 WMMAs
//   out[p] = dot(c[a],embs_c[b]) + mean_i max_j block
// ---------------------------------------------------------------------------
__global__ __launch_bounds__(256)
void pair_scores(const unsigned short* __restrict__ nb,
                 const unsigned short* __restrict__ enb,
                 const float* __restrict__ cc, const float* __restrict__ ec,
                 const int* __restrict__ fi, const int* __restrict__ si,
                 float* __restrict__ out, int P) {
    const int lane = threadIdx.x & 31;
    const int w    = threadIdx.x >> 5;
    const int p    = blockIdx.x * 8 + w;
    if (p >= P) return;                 // uniform per wave: EXEC stays all-ones

    const int a = fi[p];
    const int b = si[p];
    const int hf = lane >> 4;
    const int lr = lane & 15;

    const unsigned short* arow = nb  + ((size_t)a * KNEI + lr) * DMODEL;
    const unsigned short* brow = enb + ((size_t)b * KNEI + lr) * DMODEL;

    v8f acc = (v8f){0.f, 0.f, 0.f, 0.f, 0.f, 0.f, 0.f, 0.f};

    for (int kb = 0; kb < DMODEL; kb += 32) {
        FragB16 aF, bF;
#pragma unroll
        for (int v = 0; v < 8; ++v) {
            int k = kb + hf * 8 + ((v < 4) ? 2 * v : 16 + 2 * (v - 4));
            aF.u[v] = *(const unsigned int*)(arow + k);   // packed bf16 pair
            bF.u[v] = *(const unsigned int*)(brow + k);
        }
        acc = __builtin_amdgcn_wmma_f32_16x16x32_bf16(false, aF.v, false, bF.v,
                                                      (short)0, acc, false, false);
    }

    // VGPR r = row i (+8 in upper half), lane&15 = column j.
    float sm = 0.f;
#pragma unroll
    for (int r = 0; r < 8; ++r) {
        float m = acc[r];
        m = fmaxf(m, __shfl_xor(m, 1, 32));
        m = fmaxf(m, __shfl_xor(m, 2, 32));
        m = fmaxf(m, __shfl_xor(m, 4, 32));
        m = fmaxf(m, __shfl_xor(m, 8, 32));
        sm += m;                        // lanes<16: rows 0..7, lanes>=16: rows 8..15
    }
    sm += __shfl_xor(sm, 16, 32);
    float scores_n = sm * (1.0f / 16.0f);

    const float* ca = cc + (size_t)a * DMODEL;
    const float* eb = ec + (size_t)b * DMODEL;
    float s = 0.f;
    for (int i = lane; i < DMODEL; i += 32) s += ca[i] * eb[i];
#pragma unroll
    for (int off = 16; off > 0; off >>= 1) s += __shfl_xor(s, off, 32);

    if (lane == 0) out[p] = s + scores_n;
}

// ---------------------------------------------------------------------------
// Host-side orchestration
// ---------------------------------------------------------------------------
typedef unsigned short ush;

static inline void launch_gemm(hipStream_t st, const ush* A, const ush* B,
                               const float* bias, const float* res, float* C, ush* Ch,
                               int M, int N, int K, float alpha, int relu, bool tb) {
    dim3 grid(N / 128, M / 128), block(256);
    if (tb)
        gemm_bf16_wmma<true><<<grid, block, 0, st>>>(A, B, bias, res, C, Ch, M, N, K, alpha, relu);
    else
        gemm_bf16_wmma<false><<<grid, block, 0, st>>>(A, B, bias, res, C, Ch, M, N, K, alpha, relu);
}

static inline void launch_conv(hipStream_t st, const float* src, ush* dst, size_t n) {
    long n2 = (long)(n / 2);
    f32_to_bf16_kernel<<<(unsigned)((n2 + 255) / 256), 256, 0, st>>>(src, (unsigned int*)dst, n2);
}

extern "C" void kernel_launch(void* const* d_in, const int* in_sizes, int n_in,
                              void* d_out, int out_size, void* d_ws, size_t ws_size,
                              hipStream_t stream) {
    (void)n_in; (void)out_size; (void)ws_size;
    const int S = SEQ, DD = DMODEL, FF = FFD, NS = NSPAN;
    const int P = in_sizes[3];

    const float* x    = (const float*)d_in[0];
    const float* cmap = (const float*)d_in[1];
    const float* nmap = (const float*)d_in[2];
    const int*   fidx = (const int*)d_in[3];
    const int*   sidx = (const int*)d_in[4];
    auto F = [&](int i) { return (const float*)d_in[i]; };
    // 5..12 enc_sa (wq wk wv wo bq bk bv bo), 13..20 dec_sa, 21..28 dec_ca,
    // 29..32 enc_ff (w1 b1 w2 b2), 33..36 dec_ff, 37..50 LN g/b pairs.

    const size_t SD  = (size_t)S * DD;
    const size_t SS  = (size_t)S * S;
    const size_t ND  = (size_t)NS * DD;
    const size_t NKD = (size_t)NS * KNEI * DD;

    char* wsp = (char*)d_ws;
    auto allocF = [&](size_t n) { float* p = (float*)wsp; wsp += n * sizeof(float); return p; };
    auto allocH = [&](size_t n) { ush* p = (ush*)wsp; wsp += n * sizeof(ush); return p; };

    // f32 scratch
    float* big   = allocF(SS);      // attention logits / exp scratch
    float* t0f   = allocF(SD);      // pre-LN sums
    float* hbuf  = allocF(SD);
    float* dbuf  = allocF(SD);
    float* cbuf  = allocF(ND);
    float* ecbuf = allocF(ND);
    // bf16 operands
    ush* xh      = allocH(SD);
    ush* cmaph   = allocH((size_t)NS * S);
    ush* nmaph   = allocH((size_t)NS * KNEI * S);
    ush* t0h     = allocH(SD);
    ush* t1h     = allocH(SD);
    ush* t2h     = allocH(SD);
    ush* t3h     = allocH(SD);
    ush* bigh    = allocH(SS);      // softmax probs / FFN hidden
    ush* hbufh   = allocH(SD);
    ush* membufh = allocH(SD);
    ush* dbufh   = allocH(SD);
    ush* embsh   = allocH(SD);
    ush* nbh     = allocH(NKD);
    ush* enbh    = allocH(NKD);
    // bf16 weights
    const int widx[16] = {5, 6, 7, 8, 13, 14, 15, 16, 21, 22, 23, 24, 29, 31, 33, 35};
    ush* wh[64] = {};
    for (int i = 0; i < 16; ++i) {
        size_t n = (i < 12) ? (size_t)DD * DD : (size_t)DD * FF;
        wh[widx[i]] = allocH(n);
    }

    hipStream_t st = stream;
    const float invs = 1.0f / sqrtf((float)DD);

    // ---------------- one-time f32 -> bf16 conversions ----------------
    launch_conv(st, x, xh, SD);
    launch_conv(st, cmap, cmaph, (size_t)NS * S);
    launch_conv(st, nmap, nmaph, (size_t)NS * KNEI * S);
    for (int i = 0; i < 16; ++i) {
        size_t n = (i < 12) ? (size_t)DD * DD : (size_t)DD * FF;
        launch_conv(st, F(widx[i]), wh[widx[i]], n);
    }

    // ---------------- encoder: self-attention ----------------
    launch_gemm(st, xh, wh[5], F(9),  nullptr, nullptr, t0h, S, DD, DD, 1.f, 0, false); // Q
    launch_gemm(st, xh, wh[6], F(10), nullptr, nullptr, t1h, S, DD, DD, 1.f, 0, false); // K
    launch_gemm(st, xh, wh[7], F(11), nullptr, nullptr, t2h, S, DD, DD, 1.f, 0, false); // V
    launch_gemm(st, t0h, t1h, nullptr, nullptr, big, nullptr, S, S, DD, invs, 0, true); // QK^T
    softmax_rows<<<S, 256, 0, st>>>(big, bigh, S);
    launch_gemm(st, bigh, t2h, nullptr, nullptr, nullptr, t3h, S, DD, S, 1.f, 0, false);
    launch_gemm(st, t3h, wh[8], F(12), x, t0f, nullptr, S, DD, DD, 1.f, 0, false);      // +x
    layernorm_rows<<<S, 256, 0, st>>>(t0f, F(37), F(38), hbuf, hbufh, DD);              // enc_ln1
    // ---------------- encoder: FFN ----------------
    launch_gemm(st, hbufh, wh[29], F(30), nullptr, nullptr, bigh, S, FF, DD, 1.f, 1, false);
    launch_gemm(st, bigh, wh[31], F(32), hbuf, t0f, nullptr, S, DD, FF, 1.f, 0, false);
    layernorm_rows<<<S, 256, 0, st>>>(t0f, F(39), F(40), hbuf, hbufh, DD);              // enc_ln2
    layernorm_rows<<<S, 256, 0, st>>>(hbuf, F(41), F(42), nullptr, membufh, DD);        // enc_norm

    // ---------------- decoder: self-attention (on x) ----------------
    launch_gemm(st, xh, wh[13], F(17), nullptr, nullptr, t0h, S, DD, DD, 1.f, 0, false);
    launch_gemm(st, xh, wh[14], F(18), nullptr, nullptr, t1h, S, DD, DD, 1.f, 0, false);
    launch_gemm(st, xh, wh[15], F(19), nullptr, nullptr, t2h, S, DD, DD, 1.f, 0, false);
    launch_gemm(st, t0h, t1h, nullptr, nullptr, big, nullptr, S, S, DD, invs, 0, true);
    softmax_rows<<<S, 256, 0, st>>>(big, bigh, S);
    launch_gemm(st, bigh, t2h, nullptr, nullptr, nullptr, t3h, S, DD, S, 1.f, 0, false);
    launch_gemm(st, t3h, wh[16], F(20), x, t0f, nullptr, S, DD, DD, 1.f, 0, false);
    layernorm_rows<<<S, 256, 0, st>>>(t0f, F(43), F(44), dbuf, dbufh, DD);              // dec_ln1

    // ---------------- decoder: cross-attention (q=d, kv=mem) ----------------
    launch_gemm(st, dbufh,   wh[21], F(25), nullptr, nullptr, t0h, S, DD, DD, 1.f, 0, false);
    launch_gemm(st, membufh, wh[22], F(26), nullptr, nullptr, t1h, S, DD, DD, 1.f, 0, false);
    launch_gemm(st, membufh, wh[23], F(27), nullptr, nullptr, t2h, S, DD, DD, 1.f, 0, false);
    launch_gemm(st, t0h, t1h, nullptr, nullptr, big, nullptr, S, S, DD, invs, 0, true);
    softmax_rows<<<S, 256, 0, st>>>(big, bigh, S);
    launch_gemm(st, bigh, t2h, nullptr, nullptr, nullptr, t3h, S, DD, S, 1.f, 0, false);
    launch_gemm(st, t3h, wh[24], F(28), dbuf, t0f, nullptr, S, DD, DD, 1.f, 0, false);
    layernorm_rows<<<S, 256, 0, st>>>(t0f, F(45), F(46), dbuf, dbufh, DD);              // dec_ln2

    // ---------------- decoder: FFN ----------------
    launch_gemm(st, dbufh, wh[33], F(34), nullptr, nullptr, bigh, S, FF, DD, 1.f, 1, false);
    launch_gemm(st, bigh, wh[35], F(36), dbuf, t0f, nullptr, S, DD, FF, 1.f, 0, false);
    layernorm_rows<<<S, 256, 0, st>>>(t0f, F(47), F(48), hbuf, nullptr, DD);            // dec_ln3
    layernorm_rows<<<S, 256, 0, st>>>(hbuf, F(49), F(50), nullptr, embsh, DD);          // dec_norm

    // ---------------- span maps ----------------
    launch_gemm(st, cmaph, xh,    nullptr, nullptr, cbuf,  nullptr, NS, DD, S, 1.f, 0, false);
    launch_gemm(st, cmaph, embsh, nullptr, nullptr, ecbuf, nullptr, NS, DD, S, 1.f, 0, false);
    launch_gemm(st, nmaph, xh,    nullptr, nullptr, nullptr, nbh,  NS * KNEI, DD, S, 1.f, 0, false);
    launch_gemm(st, nmaph, embsh, nullptr, nullptr, nullptr, enbh, NS * KNEI, DD, S, 1.f, 0, false);

    // ---------------- pair scores ----------------
    pair_scores<<<(P + 7) / 8, 256, 0, st>>>(nbh, enbh, cbuf, ecbuf,
                                             fidx, sidx, (float*)d_out, P);
}
